// MultiScaleAPP_41360535061066
// MI455X (gfx1250) — matrix operand order repaired and checked by reading, not compiled, not run on hardware
//
#include <hip/hip_runtime.h>
#include <hip/hip_bf16.h>

typedef __attribute__((ext_vector_type(16))) _Float16 v16h;
typedef __attribute__((ext_vector_type(8)))  _Float16 v8h;
typedef __attribute__((ext_vector_type(4)))  _Float16 v4h;
typedef __attribute__((ext_vector_type(8)))  float    v8f;

#define Nn 100000
#define Ee 1600000
#define DIN 128
#define DOUT 64
#define NS 3

// ---------------------------------------------------------------- utilities

__global__ void f32_to_f16_kernel(const float* __restrict__ x,
                                  _Float16* __restrict__ y, long n) {
    long i = (long)blockIdx.x * blockDim.x + threadIdx.x;
    if (i < n) y[i] = (_Float16)x[i];
}

__global__ void degree_kernel(const int* __restrict__ tgt,
                              float* __restrict__ deg, int e) {
    int i = blockIdx.x * blockDim.x + threadIdx.x;
    if (i < e) atomicAdd(&deg[tgt[i]], 1.0f);
}

__global__ void deginv_kernel(const float* __restrict__ deg,
                              float* __restrict__ deg_inv,
                              float* __restrict__ selfw, int n) {
    int i = blockIdx.x * blockDim.x + threadIdx.x;
    if (i < n) {
        float d = deg[i];
        deg_inv[i] = 1.0f / fmaxf(d, 1.0f);
        selfw[i]   = (d > 0.0f) ? 1.0f : 0.0f;
    }
}

// Pack B[K x 64] (f32, row-major) into WMMA B-fragment order (f16):
// Bp[ktile][ntile][lane][j] with 16 contiguous halves per lane (32B load).
// lane: half=lane>>4 selects K 0-15 / 16-31 within the 32-K tile,
//       col = ntile*16 + (lane&15);  element j -> K = ktile*32 + half*16 + j.
__global__ void pack_b_kernel(const float* __restrict__ B,
                              _Float16* __restrict__ Bp, int K,
                              long srcStride, long dstStride) {
    const int s = blockIdx.y;
    long i = (long)blockIdx.x * blockDim.x + threadIdx.x;
    long total = (long)K * DOUT;
    if (i >= total) return;
    int j     = (int)(i & 15);
    int lane  = (int)((i >> 4) & 31);
    int nt    = (int)((i >> 9) & 3);
    int ktile = (int)(i >> 11);
    int half  = lane >> 4;
    int col   = nt * 16 + (lane & 15);
    int k     = ktile * 32 + half * 16 + j;
    Bp[(long)s * dstStride + i] = (_Float16)B[(long)s * srcStride + (long)k * DOUT + col];
}

// ---------------------------------------------------------------- WMMA GEMM
// C[s][M x 64] = A[s][M x K] (f16) * B[s][K x 64] (packed f16) + bias[s][64]
// One wave per 16x64 strip: 4 accumulators share each A fragment.
// K templated -> fully unrolled (K/32 * 4 WMMAs per wave).
template <int K>
__global__ void wmma_gemm_kernel(const _Float16* __restrict__ A,
                                 const _Float16* __restrict__ Bp,
                                 const float* __restrict__ bias,
                                 float* __restrict__ C,
                                 long aStride, long bStride,
                                 long biasStride, long cStride,
                                 int rowTiles) {
    const int lane = threadIdx.x & 31;
    const int wave = threadIdx.x >> 5;
    int tile_m = blockIdx.x * 8 + wave;
    if (tile_m >= rowTiles) tile_m = rowTiles - 1;  // benign duplicate work
    const int s = blockIdx.y;
    A    += (long)s * aStride;
    Bp   += (long)s * bStride;
    bias += (long)s * biasStride;
    C    += (long)s * cStride;

    const int half = lane >> 4;               // 0: lanes 0-15, 1: lanes 16-31
    const int row  = tile_m * 16 + (lane & 15);
    const int cb   = lane & 15;

    v8f c[4];
#pragma unroll
    for (int t = 0; t < 4; ++t) {
        float bv = bias[t * 16 + cb];
        c[t] = (v8f){bv, bv, bv, bv, bv, bv, bv, bv};
    }

#pragma unroll
    for (int kk = 0; kk < K; kk += 32) {
        // A fragment: two 128-bit loads, 16-bit A 16x32 layout.
        union { v16h v; struct { v8h lo, hi; } p; } a;
        const _Float16* ap = A + (long)row * K + kk + half * 16;
        a.p.lo = *(const v8h*)(ap);
        a.p.hi = *(const v8h*)(ap + 8);

        const _Float16* bp = Bp + (long)(kk >> 5) * 2048 + (long)lane * 16;
#pragma unroll
        for (int t = 0; t < 4; ++t) {
            v16h b = *(const v16h*)(bp + t * 512);  // 32B contiguous per lane
            c[t] = __builtin_amdgcn_wmma_f32_16x16x32_f16(
                       false, a.v, false, b, (short)0, c[t], false, false);
        }
    }

    const int mbase = tile_m * 16 + half * 8;
#pragma unroll
    for (int t = 0; t < 4; ++t)
#pragma unroll
        for (int r = 0; r < 8; ++r)
            C[(long)(mbase + r) * DOUT + t * 16 + cb] = c[t][r];
}

// ---------------------------------------------------------------- propagation
// acc[s][tgt] += in[s][src]  (float4 gathers, f32 atomics; L2-resident)
__global__ void scatter_kernel(const int* __restrict__ src,
                               const int* __restrict__ tgt,
                               const float* __restrict__ in,
                               float* __restrict__ acc, long ss) {
    long t = (long)blockIdx.x * blockDim.x + threadIdx.x;
    long e = t >> 4;
    if (e >= Ee) return;
    int q = (int)(t & 15) * 4;
    long so = (long)src[e] * DOUT + q;
    long go = (long)tgt[e] * DOUT + q;
#pragma unroll
    for (int sc = 0; sc < NS; ++sc) {
        float4 v = *(const float4*)(in + sc * ss + so);
        float* a = acc + sc * ss + go;
        atomicAdd(a + 0, v.x);
        atomicAdd(a + 1, v.y);
        atomicAdd(a + 2, v.z);
        atomicAdd(a + 3, v.w);
    }
}

// nxt = (1-t)*(in*selfw - deg_inv*acc) + t*emb ; acc = 0 (ready for next pass)
__global__ void combine_kernel(const float* __restrict__ in,
                               float* __restrict__ acc,
                               const float* __restrict__ emb,
                               const float* __restrict__ deg_inv,
                               const float* __restrict__ selfw,
                               float* __restrict__ nxt, long ss) {
    long idx = (long)blockIdx.x * blockDim.x + threadIdx.x;
    long n = idx >> 4;
    if (n >= Nn) return;
    int q = (int)(idx & 15) * 4;
    float di = deg_inv[n];
    float sw = selfw[n];
    const float tel[NS] = {0.1f, 0.2f, 0.3f};
#pragma unroll
    for (int sc = 0; sc < NS; ++sc) {
        long o = sc * ss + n * DOUT + q;
        float4 vi = *(const float4*)(in + o);
        float4 va = *(const float4*)(acc + o);
        float4 ve = *(const float4*)(emb + o);
        float t = tel[sc], omt = 1.0f - t;
        float4 r;
        r.x = omt * (vi.x * sw - di * va.x) + t * ve.x;
        r.y = omt * (vi.y * sw - di * va.y) + t * ve.y;
        r.z = omt * (vi.z * sw - di * va.z) + t * ve.z;
        r.w = omt * (vi.w * sw - di * va.w) + t * ve.w;
        *(float4*)(nxt + o) = r;
        *(float4*)(acc + o) = make_float4(0.f, 0.f, 0.f, 0.f);
    }
}

// in-place relu on f32 scales + f16 copy for the WMMA attention GEMM
__global__ void relu_conv_kernel(float* __restrict__ x,
                                 _Float16* __restrict__ xh, long total4) {
    long i = (long)blockIdx.x * blockDim.x + threadIdx.x;
    if (i >= total4) return;
    float4 v = ((float4*)x)[i];
    v.x = fmaxf(v.x, 0.f); v.y = fmaxf(v.y, 0.f);
    v.z = fmaxf(v.z, 0.f); v.w = fmaxf(v.w, 0.f);
    ((float4*)x)[i] = v;
    v4h h = {(_Float16)v.x, (_Float16)v.y, (_Float16)v.z, (_Float16)v.w};
    ((v4h*)xh)[i] = h;
}

// ---------------------------------------------------------------- attention
// one wave per node: logits[s] = <t_att[s,n], s_att[n]>, softmax over s,
// out[n] = sum_s att[s]*scales[s,n].  N = 12500 blocks * 8 waves exactly.
__global__ void attention_kernel(const float* __restrict__ scales,
                                 const float* __restrict__ t_att,
                                 const float* __restrict__ s_att,
                                 float* __restrict__ out, long ss) {
    const int lane = threadIdx.x & 31;
    const int wave = threadIdx.x >> 5;
    const long n = (long)blockIdx.x * 8 + wave;
    const long nb = n * DOUT;

    float sa0 = s_att[nb + lane];
    float sa1 = s_att[nb + lane + 32];

    float logits[NS];
#pragma unroll
    for (int sc = 0; sc < NS; ++sc) {
        long o = sc * ss + nb;
        float p = t_att[o + lane] * sa0 + t_att[o + lane + 32] * sa1;
#pragma unroll
        for (int off = 16; off > 0; off >>= 1)
            p += __shfl_xor(p, off, 32);
        logits[sc] = p;
    }
    float m  = fmaxf(logits[0], fmaxf(logits[1], logits[2]));
    float e0 = __expf(logits[0] - m);
    float e1 = __expf(logits[1] - m);
    float e2 = __expf(logits[2] - m);
    float inv = 1.0f / (e0 + e1 + e2);
    float a0 = e0 * inv, a1 = e1 * inv, a2 = e2 * inv;

    float o0 = a0 * scales[0 * ss + nb + lane]
             + a1 * scales[1 * ss + nb + lane]
             + a2 * scales[2 * ss + nb + lane];
    float o1 = a0 * scales[0 * ss + nb + lane + 32]
             + a1 * scales[1 * ss + nb + lane + 32]
             + a2 * scales[2 * ss + nb + lane + 32];
    out[nb + lane]      = o0;
    out[nb + lane + 32] = o1;
}

// ---------------------------------------------------------------- launch

extern "C" void kernel_launch(void* const* d_in, const int* in_sizes, int n_in,
                              void* d_out, int out_size, void* d_ws, size_t ws_size,
                              hipStream_t stream) {
    const float* data  = (const float*)d_in[0];
    const int*   src   = (const int*)  d_in[1];
    const int*   tgt   = (const int*)  d_in[2];
    const float* Ws    = (const float*)d_in[3];
    const float* bs    = (const float*)d_in[4];
    const float* W_src = (const float*)d_in[5];
    const float* b_src = (const float*)d_in[6];
    const float* W_tgt = (const float*)d_in[7];
    const float* b_tgt = (const float*)d_in[8];
    float* out = (float*)d_out;

    const long SS = (long)Nn * DOUT;   // per-scale feature stride (elements)
    const int  rowTiles = Nn / 16;     // 6250
    const int  gemmGrid = (rowTiles + 7) / 8;

    // workspace carve-up (256B aligned)
    size_t off = 0;
    auto carve = [&](size_t bytes) -> char* {
        char* p = (char*)d_ws + off;
        off = (off + bytes + 255) & ~(size_t)255;
        return p;
    };
    float*    deg     = (float*)   carve((size_t)Nn * 4);
    float*    deg_inv = (float*)   carve((size_t)Nn * 4);
    float*    selfw   = (float*)   carve((size_t)Nn * 4);
    _Float16* dataH   = (_Float16*)carve((size_t)Nn * DIN * 2);
    _Float16* WsP     = (_Float16*)carve((size_t)NS * DIN * DOUT * 2);
    _Float16* WsrcP   = (_Float16*)carve((size_t)DIN * DOUT * 2);
    _Float16* WtgtP   = (_Float16*)carve((size_t)DOUT * DOUT * 2);
    float*    emb     = (float*)   carve((size_t)NS * SS * 4);
    float*    buf0    = (float*)   carve((size_t)NS * SS * 4);
    float*    buf1    = (float*)   carve((size_t)NS * SS * 4);
    float*    acc     = (float*)   carve((size_t)NS * SS * 4);   // reused as t_att
    _Float16* scalesH = (_Float16*)carve((size_t)NS * SS * 2);
    float*    s_att   = (float*)   carve((size_t)SS * 4);
    (void)ws_size; (void)in_sizes; (void)n_in; (void)out_size;

    // 1. degree / normalization
    hipMemsetAsync(deg, 0, (size_t)Nn * 4, stream);
    hipMemsetAsync(acc, 0, (size_t)NS * SS * 4, stream);
    degree_kernel<<<(Ee + 255) / 256, 256, 0, stream>>>(tgt, deg, Ee);
    deginv_kernel<<<(Nn + 255) / 256, 256, 0, stream>>>(deg, deg_inv, selfw, Nn);

    // 2. f16 conversion of activations + fragment-packing of weights
    {
        long nd = (long)Nn * DIN;
        f32_to_f16_kernel<<<(nd + 255) / 256, 256, 0, stream>>>(data, dataH, nd);
        long wb = (long)DIN * DOUT;       // 8192 elements per scale
        dim3 gs((unsigned)((wb + 255) / 256), NS);
        pack_b_kernel<<<gs, 256, 0, stream>>>(Ws, WsP, DIN, wb, wb);
        pack_b_kernel<<<(wb + 255) / 256, 256, 0, stream>>>(W_src, WsrcP, DIN, 0, 0);
        long wt = (long)DOUT * DOUT;      // 4096 elements
        pack_b_kernel<<<(wt + 255) / 256, 256, 0, stream>>>(W_tgt, WtgtP, DOUT, 0, 0);
    }

    // 3. emb[s] = data @ Ws[s] + bs[s]   (WMMA, grid.y = scales)
    {
        dim3 g(gemmGrid, NS);
        wmma_gemm_kernel<DIN><<<g, 256, 0, stream>>>(
            dataH, WsP, bs, emb, 0, (long)DIN * DOUT, DOUT, SS, rowTiles);
    }

    // 4. 10 propagation iterations, ping-pong buf0/buf1, acc self-rezeroed
    const float* cur = emb;
    float* nxt = buf0;
    for (int it = 0; it < 10; ++it) {
        scatter_kernel<<<(int)(((long)Ee * 16 + 255) / 256), 256, 0, stream>>>(
            src, tgt, cur, acc, SS);
        combine_kernel<<<(int)(((long)Nn * 16 + 255) / 256), 256, 0, stream>>>(
            cur, acc, emb, deg_inv, selfw, nxt, SS);
        cur = nxt;
        nxt = (nxt == buf0) ? buf1 : buf0;
    }
    float* scales = (float*)cur;   // final pre-relu propagation result

    // 5. relu (in place) + f16 copy
    {
        long total4 = (long)NS * SS / 4;
        relu_conv_kernel<<<(int)((total4 + 255) / 256), 256, 0, stream>>>(
            scales, scalesH, total4);
    }

    // 6. s_att = data @ W_src + b_src   (WMMA)
    {
        dim3 g(gemmGrid, 1);
        wmma_gemm_kernel<DIN><<<g, 256, 0, stream>>>(
            dataH, WsrcP, b_src, s_att, 0, 0, 0, 0, rowTiles);
    }

    // 7. t_att[s] = scales[s] @ W_tgt + b_tgt   (WMMA, K=64; reuse acc)
    float* t_att = acc;
    {
        dim3 g(gemmGrid, NS);
        wmma_gemm_kernel<DOUT><<<g, 256, 0, stream>>>(
            scalesH, WtgtP, b_tgt, t_att, SS, 0, 0, SS, rowTiles);
    }

    // 8. fused logits -> softmax over scales -> weighted sum
    attention_kernel<<<Nn / 8, 256, 0, stream>>>(scales, t_att, s_att, out, SS);
}
